// StandardAttention_2310692405603
// MI455X (gfx1250) — matrix operand order repaired and checked by reading, not compiled, not run on hardware
//
#include <hip/hip_runtime.h>
#include <math.h>

// ---------------------------------------------------------------------------
// MHA forward for MI455X (gfx1250, wave32, WMMA).
// B=4, S=2048, D_MODEL=2048, H=16, Dh=128.
// Precision: bf16 inputs to V_WMMA_F32_16X16X32_BF16, f32 accumulate.
// ---------------------------------------------------------------------------

#define S_LEN   2048
#define DMODEL  2048
#define NHEADS  16
#define DHEAD   128
#define BATCH   4
#define MROWS   (BATCH * S_LEN)      // 8192

typedef __attribute__((ext_vector_type(16))) __bf16 v16bf;
typedef __attribute__((ext_vector_type(8)))  float  v8f;

union Frag {
    v16bf v;
    uint4 u4[2];
    unsigned short s[16];
};

__device__ __forceinline__ unsigned short f2bf_u16(float f) {
    union { float f; unsigned u; } v; v.f = f;
    unsigned u = v.u;
    u += 0x7FFFu + ((u >> 16) & 1u);   // round-to-nearest-even
    return (unsigned short)(u >> 16);
}

__device__ __forceinline__ v8f vzero8() {
    v8f z = {0.f, 0.f, 0.f, 0.f, 0.f, 0.f, 0.f, 0.f};
    return z;
}

__device__ __forceinline__ v8f wmma_bf16(v16bf a, v16bf b, v8f c) {
    // (neg_a, A, neg_b, B, c_mod, C, reuse_a, reuse_b)
    return __builtin_amdgcn_wmma_f32_16x16x32_bf16(false, a, false, b,
                                                   (short)0, c, false, false);
}

// ---------------------------------------------------------------------------
// f32 -> bf16 elementwise convert (grid-stride)
// ---------------------------------------------------------------------------
__global__ void cvt_f32_bf16(const float* __restrict__ src,
                             unsigned short* __restrict__ dst, int n) {
    int i = blockIdx.x * blockDim.x + threadIdx.x;
    int stride = gridDim.x * blockDim.x;
    for (; i < n; i += stride) dst[i] = f2bf_u16(src[i]);
}

// ---------------------------------------------------------------------------
// GEMM: Y[m,n] = sum_k A[m,k] * W[n,k]   (A: [8192,2048] bf16, W: [2048,2048] bf16)
// One wave computes a 64x64 tile (4x4 WMMA tiles): 16 WMMA per 16 b128 loads
// per K-step of 32 (AI ~= 32 flop/byte from L2; both operands L2-resident).
// mode 0: store bf16 as [b][h][s][d]      (Q, K for attention)
// mode 1: store bf16 as [b][h][d][s]      (V transposed for attention)
// mode 2: store f32 row-major [m][n]      (final output projection)
// ---------------------------------------------------------------------------
__global__ __launch_bounds__(256) void gemm_proj(
        const unsigned short* __restrict__ A,
        const unsigned short* __restrict__ W,
        unsigned short* __restrict__ outb,
        float* __restrict__ outf, int mode) {
    const int lane = threadIdx.x & 31;
    const int wid  = threadIdx.x >> 5;
    const int gw   = blockIdx.x * 8 + wid;      // 0 .. 4095 (exact)
    const int nb   = gw & 31;                   // 32 n-blocks of 64
    const int mb   = gw >> 5;                   // 128 m-blocks of 64
    const int lh   = lane >> 4;                 // lane half (0/1)
    const int ln   = lane & 15;

    // Per-lane row base pointers (A rows m, W rows n)
    const unsigned short* arow[4];
    const unsigned short* wrow[4];
    #pragma unroll
    for (int i = 0; i < 4; ++i) {
        arow[i] = A + (size_t)(mb * 64 + 16 * i + ln) * DMODEL;
        wrow[i] = W + (size_t)(nb * 64 + 16 * i + ln) * DMODEL;
    }

    v8f acc[4][4];
    #pragma unroll
    for (int i = 0; i < 4; ++i)
        #pragma unroll
        for (int j = 0; j < 4; ++j) acc[i][j] = vzero8();

    const int PF = 256;                         // prefetch 8 K-steps ahead
    for (int k0 = 0; k0 < DMODEL; k0 += 32) {
        if (k0 + PF < DMODEL) {                 // uniform branch
            #pragma unroll
            for (int i = 0; i < 4; ++i) {
                __builtin_prefetch(arow[i] + k0 + PF, 0, 1);
                __builtin_prefetch(wrow[i] + k0 + PF, 0, 1);
            }
        }
        Frag a[4], b[4];
        #pragma unroll
        for (int i = 0; i < 4; ++i) {
            // A-fragment (16x32 MxK): K split {e+8lh, 16+e+8lh}
            const unsigned short* ap = arow[i] + k0 + 8 * lh;
            a[i].u4[0] = *(const uint4*)(ap);
            a[i].u4[1] = *(const uint4*)(ap + 16);
            // B-fragment (32x16 KxN): lane col n = row of W, K split {e+8lh, 16+e+8lh}
            const unsigned short* wp = wrow[i] + k0 + 8 * lh;
            b[i].u4[0] = *(const uint4*)(wp);
            b[i].u4[1] = *(const uint4*)(wp + 16);
        }
        #pragma unroll
        for (int i = 0; i < 4; ++i)
            #pragma unroll
            for (int j = 0; j < 4; ++j)
                acc[i][j] = wmma_bf16(a[i].v, b[j].v, acc[i][j]);
    }

    // C/D layout: lane holds column n = ln, rows m = r + 8*lh
    #pragma unroll
    for (int i = 0; i < 4; ++i) {
        #pragma unroll
        for (int j = 0; j < 4; ++j) {
            #pragma unroll
            for (int r = 0; r < 8; ++r) {
                const int m = mb * 64 + 16 * i + r + 8 * lh;
                const int n = nb * 64 + 16 * j + ln;
                if (mode == 2) {
                    outf[(size_t)m * DMODEL + n] = acc[i][j][r];
                } else {
                    const int bidx = m >> 11, s = m & (S_LEN - 1);
                    const int h = n >> 7, d = n & (DHEAD - 1);
                    const unsigned short val = f2bf_u16(acc[i][j][r]);
                    if (mode == 0)
                        outb[(((size_t)(bidx * NHEADS + h)) * S_LEN + s) * DHEAD + d] = val;
                    else
                        outb[(((size_t)(bidx * NHEADS + h)) * DHEAD + d) * S_LEN + s] = val;
                }
            }
        }
    }
}

// ---------------------------------------------------------------------------
// Causal flash attention. One wave handles one (b, h, 16-query tile).
// Computes S^T = K @ Q^T so that the softmaxed C-tile is directly the
// A-fragment for O = P @ V (no cross-lane transpose needed).
// Q: [b][h][s][d] bf16, K: [b][h][t][d] bf16, V: [b][h][d][t] bf16 (transposed)
// out: attn merged heads [b][s][h*128+d] bf16
// ---------------------------------------------------------------------------
__global__ __launch_bounds__(256) void flash_attn(
        const unsigned short* __restrict__ Qh,
        const unsigned short* __restrict__ Kh,
        const unsigned short* __restrict__ Vt,
        unsigned short* __restrict__ attn) {
    const int lane = threadIdx.x & 31;
    const int wid  = threadIdx.x >> 5;
    const int gw   = blockIdx.x * 8 + wid;   // 0 .. 8191 (exact)
    const int qt   = gw & 127;               // query tile (16 rows)
    const int bh   = gw >> 7;                // b*16 + h
    const int b    = bh >> 4;
    const int h    = bh & 15;
    const int lh   = lane >> 4;
    const int ln   = lane & 15;
    const int qrow = qt * 16 + ln;           // this lane's query row

    const unsigned short* Qbase = Qh + (size_t)bh * S_LEN * DHEAD;
    const unsigned short* Kbase = Kh + (size_t)bh * S_LEN * DHEAD;
    const unsigned short* Vbase = Vt + (size_t)bh * DHEAD * S_LEN;

    // Q as B-fragments (Q^T is 128x16 KxN): lane col = query ln, K = d
    Frag qf[4];
    {
        const unsigned short* qp = Qbase + (size_t)qrow * DHEAD + 8 * lh;
        #pragma unroll
        for (int c = 0; c < 4; ++c) {
            qf[c].u4[0] = *(const uint4*)(qp + 32 * c);
            qf[c].u4[1] = *(const uint4*)(qp + 32 * c + 16);
        }
    }

    v8f acc[8];
    #pragma unroll
    for (int j = 0; j < 8; ++j) acc[j] = vzero8();
    float m_i = -INFINITY, l_i = 0.f;
    const float scale = 0.08838834764831845f;   // 1/sqrt(128)

    const int t_end = qt * 16 + 16;             // last valid key + 1
    for (int t0 = 0; t0 < t_end; t0 += 32) {
        if (t0 + 32 < t_end) {                  // uniform: prefetch next K tile
            __builtin_prefetch(Kbase + (size_t)(t0 + 32 + ln) * DHEAD, 0, 1);
            __builtin_prefetch(Kbase + (size_t)(t0 + 48 + ln) * DHEAD, 0, 1);
        }
        // ---- S^T tiles: s0 = keys t0..t0+15, s1 = keys t0+16..t0+31 ----
        v8f s0 = vzero8(), s1 = vzero8();
        #pragma unroll
        for (int c = 0; c < 4; ++c) {
            Frag ak0, ak1;
            const unsigned short* kp = Kbase + 32 * c + 8 * lh;
            const unsigned short* p0 = kp + (size_t)(t0 + ln) * DHEAD;
            const unsigned short* p1 = kp + (size_t)(t0 + 16 + ln) * DHEAD;
            ak0.u4[0] = *(const uint4*)(p0);
            ak0.u4[1] = *(const uint4*)(p0 + 16);
            ak1.u4[0] = *(const uint4*)(p1);
            ak1.u4[1] = *(const uint4*)(p1 + 16);
            s0 = wmma_bf16(ak0.v, qf[c].v, s0);
            s1 = wmma_bf16(ak1.v, qf[c].v, s1);
        }

        // ---- scale + causal mask (lane holds query qrow; keys r+8lh) ----
        float p0v[8], p1v[8];
        float mx = -INFINITY;
        #pragma unroll
        for (int r = 0; r < 8; ++r) {
            const int t = t0 + r + 8 * lh;
            const float x0 = (t      <= qrow) ? s0[r] * scale : -INFINITY;
            const float x1 = (t + 16 <= qrow) ? s1[r] * scale : -INFINITY;
            p0v[r] = x0; p1v[r] = x1;
            mx = fmaxf(mx, fmaxf(x0, x1));
        }
        mx = fmaxf(mx, __shfl_xor(mx, 16, 32));
        const float m_new = fmaxf(m_i, mx);
        const float rescale = __expf(m_i - m_new);

        Frag ap;
        float rs = 0.f;
        #pragma unroll
        for (int r = 0; r < 8; ++r) {
            const float e0 = __expf(p0v[r] - m_new);
            const float e1 = __expf(p1v[r] - m_new);
            rs += e0 + e1;
            ap.s[r]     = f2bf_u16(e0);   // A-frag elems 0-7  <- keys t0 +  r+8lh
            ap.s[r + 8] = f2bf_u16(e1);   // A-frag elems 8-15 <- keys t0+16+r+8lh
        }
        rs += __shfl_xor(rs, 16, 32);
        l_i = l_i * rescale + rs;
        m_i = m_new;

        // ---- rescale O accumulator (rows of O live across VGPRs) ----
        float rscv[8];
        #pragma unroll
        for (int r = 0; r < 8; ++r)
            rscv[r] = __shfl(rescale, r + 8 * lh, 32);
        #pragma unroll
        for (int j = 0; j < 8; ++j)
            #pragma unroll
            for (int r = 0; r < 8; ++r)
                acc[j][r] *= rscv[r];

        // ---- O += P @ V  (V pre-transposed: contiguous B-fragments) ----
        #pragma unroll
        for (int j = 0; j < 8; ++j) {
            Frag bv;
            const unsigned short* vp =
                Vbase + (size_t)(16 * j + ln) * S_LEN + t0 + 8 * lh;
            bv.u4[0] = *(const uint4*)(vp);
            bv.u4[1] = *(const uint4*)(vp + 16);
            acc[j] = wmma_bf16(ap.v, bv.v, acc[j]);
        }
    }

    // ---- normalize and store merged heads [b][s][h*128+d] bf16 ----
    float invl[8];
    #pragma unroll
    for (int r = 0; r < 8; ++r)
        invl[r] = 1.f / __shfl(l_i, r + 8 * lh, 32);
    #pragma unroll
    for (int j = 0; j < 8; ++j) {
        #pragma unroll
        for (int r = 0; r < 8; ++r) {
            const int q = qt * 16 + r + 8 * lh;
            const int n = h * DHEAD + 16 * j + ln;
            attn[((size_t)b * S_LEN + q) * DMODEL + n] =
                f2bf_u16(acc[j][r] * invl[r]);
        }
    }
}

// ---------------------------------------------------------------------------
// Launcher
// ---------------------------------------------------------------------------
extern "C" void kernel_launch(void* const* d_in, const int* in_sizes, int n_in,
                              void* d_out, int out_size, void* d_ws, size_t ws_size,
                              hipStream_t stream) {
    const float* x  = (const float*)d_in[0];
    const float* Wq = (const float*)d_in[1];
    const float* Wk = (const float*)d_in[2];
    const float* Wv = (const float*)d_in[3];
    const float* Wo = (const float*)d_in[4];
    float* out = (float*)d_out;

    const size_t NX = (size_t)MROWS * DMODEL;   // 16,777,216 elems (x, Q, K, V, attn)
    const size_t NW = (size_t)DMODEL * DMODEL;  //  4,194,304 elems per weight

    char* ws = (char*)d_ws;
    size_t off = 0;
    auto carve = [&](size_t bytes) -> unsigned short* {
        unsigned short* p = (unsigned short*)(ws + off);
        off += (bytes + 255) & ~(size_t)255;
        return p;
    };
    unsigned short* xb   = carve(NX * 2);
    unsigned short* wqb  = carve(NW * 2);
    unsigned short* wkb  = carve(NW * 2);
    unsigned short* wvb  = carve(NW * 2);
    unsigned short* wob  = carve(NW * 2);
    unsigned short* Qh   = carve(NX * 2);   // [b][h][s][d]
    unsigned short* Kh   = carve(NX * 2);   // [b][h][t][d]
    unsigned short* Vt   = carve(NX * 2);   // [b][h][d][t]
    unsigned short* attn = carve(NX * 2);   // [b][s][h*128+d]

    // f32 -> bf16
    cvt_f32_bf16<<<4096, 256, 0, stream>>>(x,  xb,  (int)NX);
    cvt_f32_bf16<<<1024, 256, 0, stream>>>(Wq, wqb, (int)NW);
    cvt_f32_bf16<<<1024, 256, 0, stream>>>(Wk, wkb, (int)NW);
    cvt_f32_bf16<<<1024, 256, 0, stream>>>(Wv, wvb, (int)NW);
    cvt_f32_bf16<<<1024, 256, 0, stream>>>(Wo, wob, (int)NW);

    // projections: 4096 waves (128 m-blocks x 32 n-blocks) -> 512 blocks
    gemm_proj<<<512, 256, 0, stream>>>(xb, wqb, Qh, nullptr, 0);
    gemm_proj<<<512, 256, 0, stream>>>(xb, wkb, Kh, nullptr, 0);
    gemm_proj<<<512, 256, 0, stream>>>(xb, wvb, Vt, nullptr, 1);

    // attention: 8192 waves -> 1024 blocks of 8 waves
    flash_attn<<<1024, 256, 0, stream>>>(Qh, Kh, Vt, attn);

    // output projection -> f32 d_out
    gemm_proj<<<512, 256, 0, stream>>>(attn, wob, nullptr, out, 2);
}
